// Attention_26663156974296
// MI455X (gfx1250) — matrix operand order repaired and checked by reading, not compiled, not run on hardware
//
#include <hip/hip_runtime.h>

// ---------------------------------------------------------------------------
// Attention block for MI455X (gfx1250).
//   0) fp32 -> f16 conversion of x / w_qkv / w_out
//   1) QKV GEMM  (f16 WMMA, double-buffered LDS, async-to-LDS staging)
//   2) RoPE + repack (Qp, K^T, Vp)
//   3) flash attention (WMMA QK^T + online softmax with WMMA row-sums + WMMA PV)
//   4) output GEMM
// ---------------------------------------------------------------------------

typedef __attribute__((ext_vector_type(16))) _Float16 v16h;
typedef __attribute__((ext_vector_type(8)))  _Float16 v8h;
typedef __attribute__((ext_vector_type(2)))  _Float16 v2h;
typedef __attribute__((ext_vector_type(8)))  float    v8f;
typedef __attribute__((ext_vector_type(4)))  float    v4f;
typedef __attribute__((ext_vector_type(4)))  int      v4i;

static constexpr int BATCH  = 2;
static constexpr int SEQ    = 2048;
static constexpr int DMODEL = 1024;
static constexpr int NH     = 16;
static constexpr int HD     = 64;
static constexpr int MROWS  = BATCH * SEQ;   // 4096
static constexpr int NQKV   = 3 * NH * HD;   // 3072

#define WMMA_F16(a, b, c) \
  __builtin_amdgcn_wmma_f32_16x16x32_f16(false, (a), false, (b), (short)0, (c), false, false)

// ---------------------------------------------------------------------------
// Async global -> LDS staging (gfx1250 GLOBAL_LOAD_ASYNC_TO_LDS_B128),
// guarded so we fall back to plain 16B vector copies if the builtin is absent.
// The builtin takes generic int4* pointers (per clang diagnostic).
// ---------------------------------------------------------------------------
#if defined(__has_builtin)
#if __has_builtin(__builtin_amdgcn_global_load_async_to_lds_b128)
#define ASYNC_LDS 1
#endif
#endif
#ifndef ASYNC_LDS
#define ASYNC_LDS 0
#endif

__device__ __forceinline__ void copy16_to_lds(const _Float16* g, _Float16* l) {
#if ASYNC_LDS
  __builtin_amdgcn_global_load_async_to_lds_b128((v4i*)(void*)g, (v4i*)(void*)l, 0, 0);
#else
  *(v8h*)l = *(const v8h*)g;
#endif
}

__device__ __forceinline__ void lds_copy_wait() {
#if ASYNC_LDS
#if __has_builtin(__builtin_amdgcn_s_wait_asynccnt)
  __builtin_amdgcn_s_wait_asynccnt(0);
#else
  asm volatile("s_wait_asynccnt 0" ::: "memory");
#endif
#endif
}

// ---------------------------------------------------------------------------
// WMMA fragment helpers (layouts per ISA 7.12.2)
// ---------------------------------------------------------------------------
__device__ __forceinline__ int kmap2(int t, int g) {
  return (t < 4) ? (8 * g + 2 * t) : (16 + 8 * g + 2 * (t - 4));
}

// A fragment (16x32 f16): lane%16 = M row, interleaved K pairs per lane group.
template <typename PT>
__device__ __forceinline__ v16h load_a_frag(PT row, int g) {
  v16h a;
#pragma unroll
  for (int t = 0; t < 8; ++t) {
    int k = kmap2(t, g);
    v2h p = *(const v2h*)(&row[k]);
    a[2 * t]     = p[0];
    a[2 * t + 1] = p[1];
  }
  return a;
}

// B fragment (32x16 f16): lane%16 = N column, elements = K rows (table order, row stride ld).
template <typename PT>
__device__ __forceinline__ v16h load_b_frag(PT p, int g, int ld) {
  v16h b;
#pragma unroll
  for (int j = 0; j < 16; ++j) {
    int k = (j < 8) ? (8 * g + j) : (16 + 8 * g + (j - 8));
    b[j] = p[(size_t)k * ld];
  }
  return b;
}

// ---------------------------------------------------------------------------
// 0) fp32 -> f16 elementwise conversion (8 elements / thread)
// ---------------------------------------------------------------------------
__global__ __launch_bounds__(256) void cvt_f32_f16_kernel(
    const float* __restrict__ in, _Float16* __restrict__ out) {
  size_t i = ((size_t)blockIdx.x * blockDim.x + threadIdx.x) * 8;
  v4f f0 = *(const v4f*)(in + i);
  v4f f1 = *(const v4f*)(in + i + 4);
  v8h h;
#pragma unroll
  for (int j = 0; j < 4; ++j) { h[j] = (_Float16)f0[j]; h[4 + j] = (_Float16)f1[j]; }
  *(v8h*)(out + i) = h;
}

// ---------------------------------------------------------------------------
// 1/4) f16 GEMM: C[M,N] = A[M,K] * B[K,N].  Block tile 64x128, BK=64,
// double-buffered LDS with async staging.  8 waves, 16x64 C band per wave.
// ---------------------------------------------------------------------------
static constexpr int BK     = 64;
static constexpr int LDA_S  = 72;    // 64 + 8 pad halves (144B rows, 16B aligned)
static constexpr int LDB_S  = 136;   // 128 + 8 pad halves (272B rows, 16B aligned)

template <bool STORE_HALF>
__global__ __launch_bounds__(256) void gemm_f16_kernel(
    const _Float16* __restrict__ A, const _Float16* __restrict__ Bw,
    void* __restrict__ Cp, int M, int N, int K) {
  __shared__ _Float16 lds_a[2][64 * LDA_S];
  __shared__ _Float16 lds_b[2][BK * LDB_S];

  const int tid  = threadIdx.x;
  const int wave = tid >> 5;
  const int lane = tid & 31;
  const int g    = lane >> 4;
  const int ln   = lane & 15;

  const int m0 = blockIdx.y * 64;
  const int n0 = blockIdx.x * 128;
  const int wm = (wave & 3) * 16;
  const int wn = (wave >> 2) * 64;

  auto stage = [&](int buf, int kk) {
    // A tile: 64 rows x 64 halves = 512 x 16B chunks, 2 per thread
#pragma unroll
    for (int it = 0; it < 2; ++it) {
      int c = tid + it * 256;
      int r = c >> 3, col = (c & 7) * 8;
      copy16_to_lds(A + (size_t)(m0 + r) * K + kk + col,
                    &lds_a[buf][r * LDA_S + col]);
    }
    // B tile: 64 rows x 128 halves = 1024 x 16B chunks, 4 per thread
#pragma unroll
    for (int it = 0; it < 4; ++it) {
      int c = tid + it * 256;
      int r = c >> 4, col = (c & 15) * 8;
      copy16_to_lds(Bw + (size_t)(kk + r) * N + n0 + col,
                    &lds_b[buf][r * LDB_S + col]);
    }
  };

  v8f zf = {};
  v8f acc[4] = {zf, zf, zf, zf};

  stage(0, 0);
  lds_copy_wait();
  __syncthreads();

  const int nsteps = K / BK;
  for (int s = 0; s < nsteps; ++s) {
    const int buf = s & 1;
    if (s + 1 < nsteps) stage(buf ^ 1, (s + 1) * BK);

#pragma unroll
    for (int kc = 0; kc < 2; ++kc) {
      v16h af = load_a_frag(&lds_a[buf][(wm + ln) * LDA_S + kc * 32], g);
#pragma unroll
      for (int c = 0; c < 4; ++c) {
        v16h bf = load_b_frag(&lds_b[buf][(kc * 32) * LDB_S + wn + c * 16 + ln], g, LDB_S);
        acc[c] = WMMA_F16(af, bf, acc[c]);
      }
    }
    lds_copy_wait();
    __syncthreads();
  }

  // C/D layout: lane%16 = N col, VGPR r -> M row r + 8g
#pragma unroll
  for (int c = 0; c < 4; ++c) {
#pragma unroll
    for (int r = 0; r < 8; ++r) {
      int row = m0 + wm + r + 8 * g;
      int col = n0 + wn + c * 16 + ln;
      if (STORE_HALF)
        ((_Float16*)Cp)[(size_t)row * N + col] = (_Float16)acc[c][r];
      else
        ((float*)Cp)[(size_t)row * N + col] = acc[c][r];
    }
  }
}

// ---------------------------------------------------------------------------
// 2) RoPE + repack.  qkv f16 [B,L,3H*HD] ->
//    Qp [B,H,L,HD] (rope * HD^-0.5), Kt [B,H,HD,L] (rope), Vp [B,H,L,HD]
// ---------------------------------------------------------------------------
__global__ __launch_bounds__(256) void rope_pack_kernel(
    const _Float16* __restrict__ qkv, const float* __restrict__ cosT,
    const float* __restrict__ sinT, _Float16* __restrict__ Qp,
    _Float16* __restrict__ Kt, _Float16* __restrict__ Vp) {
  int tid = blockIdx.x * blockDim.x + threadIdx.x;
  int d = tid & 31;
  int l = (tid >> 5) & (SEQ - 1);
  int h = (tid >> 16) & (NH - 1);
  int b = tid >> 20;

  const _Float16* row = qkv + (size_t)(b * SEQ + l) * NQKV;
  float c = cosT[l * HD + d];
  float s = sinT[l * HD + d];

  float q0 = (float)row[h * HD + d];
  float q1 = (float)row[h * HD + d + 32];
  float k0 = (float)row[(NH + h) * HD + d];
  float k1 = (float)row[(NH + h) * HD + d + 32];
  float v0 = (float)row[(2 * NH + h) * HD + d];
  float v1 = (float)row[(2 * NH + h) * HD + d + 32];

  const float scale = 0.125f;  // HD^-0.5 folded into Q
  float qr0 = (q0 * c - q1 * s) * scale;
  float qr1 = (q1 * c + q0 * s) * scale;
  float kr0 = k0 * c - k1 * s;
  float kr1 = k1 * c + k0 * s;

  int bh = b * NH + h;
  _Float16* qo = Qp + ((size_t)bh * SEQ + l) * HD + d;
  qo[0]  = (_Float16)qr0;
  qo[32] = (_Float16)qr1;
  _Float16* vo = Vp + ((size_t)bh * SEQ + l) * HD + d;
  vo[0]  = (_Float16)v0;
  vo[32] = (_Float16)v1;
  _Float16* ko = Kt + ((size_t)bh * HD + d) * SEQ + l;
  ko[0]        = (_Float16)kr0;
  ko[32 * SEQ] = (_Float16)kr1;
}

// ---------------------------------------------------------------------------
// 3) Flash attention.  Block = (b,h, 64-query tile), 4 waves x 16 query rows.
//    64 keys / iteration: S(16x64) via 8 WMMAs, online softmax (WMMA row-sums,
//    shuffle row-max), O(16x64) += P V via 8 WMMAs + 2 WMMAs for row sums.
// ---------------------------------------------------------------------------
static constexpr int PLDS_S = 68;  // 64 + 4 pad halves

__global__ __launch_bounds__(128) void attn_kernel(
    const _Float16* __restrict__ Qp, const _Float16* __restrict__ Kt,
    const _Float16* __restrict__ Vp, _Float16* __restrict__ Og) {
  __shared__ _Float16 p_lds[4][16 * PLDS_S];

  const int wave = threadIdx.x >> 5;
  const int lane = threadIdx.x & 31;
  const int g    = lane >> 4;
  const int ln   = lane & 15;

  const int bh = blockIdx.y;  // b*NH + h
  const int b  = bh >> 4;
  const int h  = bh & (NH - 1);
  const int q0 = blockIdx.x * 64 + wave * 16;

  const _Float16* Qb  = Qp + ((size_t)bh * SEQ + q0) * HD;
  const _Float16* Ktb = Kt + (size_t)bh * HD * SEQ;
  const _Float16* Vb  = Vp + (size_t)bh * SEQ * HD;

  // Q A-fragments for K-chunks d=0..31 and d=32..63
  v16h aq[2];
#pragma unroll
  for (int c2 = 0; c2 < 2; ++c2)
    aq[c2] = load_a_frag(Qb + ln * HD + c2 * 32, g);

  v16h ones;
#pragma unroll
  for (int j = 0; j < 16; ++j) ones[j] = (_Float16)1.0f;

  v8f zf = {};
  v8f o[4] = {zf, zf, zf, zf};
  float mrun[8], lrun[8];
#pragma unroll
  for (int r = 0; r < 8; ++r) { mrun[r] = -1e30f; lrun[r] = 0.f; }

  _Float16* myp = p_lds[wave];

  for (int kb = 0; kb < SEQ; kb += 64) {
    // ---- S = Q K^T for 64 keys: four 16x16 D tiles ----
    v8f sv[4] = {zf, zf, zf, zf};
#pragma unroll
    for (int c2 = 0; c2 < 2; ++c2) {
      const _Float16* kp = Ktb + (size_t)(c2 * 32) * SEQ + kb + ln;
#pragma unroll
      for (int t = 0; t < 4; ++t) {
        v16h bk = load_b_frag(kp + 16 * t, g, SEQ);  // keys kb+16t .. kb+16t+15
        sv[t] = WMMA_F16(aq[c2], bk, sv[t]);
      }
    }

    // ---- online softmax: row max via shuffles, exp, stage P to LDS ----
    float fscale[8];
#pragma unroll
    for (int r = 0; r < 8; ++r) {
      float mx = fmaxf(fmaxf(sv[0][r], sv[1][r]), fmaxf(sv[2][r], sv[3][r]));
#pragma unroll
      for (int off = 8; off; off >>= 1) mx = fmaxf(mx, __shfl_xor(mx, off, 16));
      float mnew = fmaxf(mrun[r], mx);
      fscale[r]  = __expf(mrun[r] - mnew);
      mrun[r]    = mnew;
      int m = r + 8 * g;
#pragma unroll
      for (int t = 0; t < 4; ++t)
        myp[m * PLDS_S + 16 * t + ln] = (_Float16)__expf(sv[t][r] - mnew);
    }
#pragma unroll
    for (int c = 0; c < 4; ++c)
#pragma unroll
      for (int r = 0; r < 8; ++r) o[c][r] *= fscale[r];

    // ---- P (D-layout) -> A-fragments via per-wave LDS (same-wave DS ops
    //      are in-order; no barrier needed) ----
    v16h pa0 = load_a_frag(&myp[ln * PLDS_S], g);        // keys kb..kb+31
    v16h pa1 = load_a_frag(&myp[ln * PLDS_S + 32], g);   // keys kb+32..kb+63

    // ---- row sums of P via WMMA against ones (replaces shuffle tree) ----
    v8f lsum = WMMA_F16(pa0, ones, zf);
    lsum = WMMA_F16(pa1, ones, lsum);
#pragma unroll
    for (int r = 0; r < 8; ++r) lrun[r] = lrun[r] * fscale[r] + lsum[r];

    // ---- O += P V ----
#pragma unroll
    for (int c = 0; c < 4; ++c) {
      v16h bv0 = load_b_frag(Vb + (size_t)kb * HD + c * 16 + ln, g, HD);
      v16h bv1 = load_b_frag(Vb + (size_t)(kb + 32) * HD + c * 16 + ln, g, HD);
      o[c] = WMMA_F16(pa0, bv0, o[c]);
      o[c] = WMMA_F16(pa1, bv1, o[c]);
    }
  }

  // ---- finalize: divide by row sums, store O as f16 [B*L, H*HD] ----
#pragma unroll
  for (int r = 0; r < 8; ++r) {
    float inv = 1.0f / lrun[r];
    int row = b * SEQ + q0 + r + 8 * g;
#pragma unroll
    for (int c = 0; c < 4; ++c) {
      int col = h * HD + c * 16 + ln;
      Og[(size_t)row * (NH * HD) + col] = (_Float16)(o[c][r] * inv);
    }
  }
}

// ---------------------------------------------------------------------------
// Launcher.  Inputs: x, cos, sin, w_qkv, w_out (fp32).  Output fp32.
// Workspace (f16): x16 | wqkv16 | wout16 | qkv | Qp | Kt | Vp | Og
// ---------------------------------------------------------------------------
extern "C" void kernel_launch(void* const* d_in, const int* in_sizes, int n_in,
                              void* d_out, int out_size, void* d_ws, size_t ws_size,
                              hipStream_t stream) {
  const float* x     = (const float*)d_in[0];
  const float* cosT  = (const float*)d_in[1];
  const float* sinT  = (const float*)d_in[2];
  const float* w_qkv = (const float*)d_in[3];
  const float* w_out = (const float*)d_in[4];

  char* ws = (char*)d_ws;
  const size_t SZ_X   = (size_t)MROWS * DMODEL * sizeof(_Float16);          // 8.4 MB
  const size_t SZ_WQ  = (size_t)DMODEL * NQKV * sizeof(_Float16);           // 6.3 MB
  const size_t SZ_WO  = (size_t)DMODEL * DMODEL * sizeof(_Float16);         // 2.1 MB
  const size_t SZ_QKV = (size_t)MROWS * NQKV * sizeof(_Float16);            // 25.2 MB
  const size_t SZ_HT  = (size_t)BATCH * NH * SEQ * HD * sizeof(_Float16);   // 8.4 MB

  _Float16* x16    = (_Float16*)(ws);
  _Float16* wqkv16 = (_Float16*)(ws + SZ_X);
  _Float16* wout16 = (_Float16*)(ws + SZ_X + SZ_WQ);
  _Float16* qkv    = (_Float16*)(ws + SZ_X + SZ_WQ + SZ_WO);
  _Float16* Qp     = (_Float16*)(ws + SZ_X + SZ_WQ + SZ_WO + SZ_QKV);
  _Float16* Ktw    = (_Float16*)(ws + SZ_X + SZ_WQ + SZ_WO + SZ_QKV + SZ_HT);
  _Float16* Vp     = (_Float16*)(ws + SZ_X + SZ_WQ + SZ_WO + SZ_QKV + 2 * SZ_HT);
  _Float16* Og     = (_Float16*)(ws + SZ_X + SZ_WQ + SZ_WO + SZ_QKV + 3 * SZ_HT);

  // 0) fp32 -> f16 conversions
  cvt_f32_f16_kernel<<<(MROWS * DMODEL) / (256 * 8), 256, 0, stream>>>(x, x16);
  cvt_f32_f16_kernel<<<(DMODEL * NQKV) / (256 * 8), 256, 0, stream>>>(w_qkv, wqkv16);
  cvt_f32_f16_kernel<<<(DMODEL * DMODEL) / (256 * 8), 256, 0, stream>>>(w_out, wout16);

  // 1) qkv = x @ w_qkv   (4096 x 3072 x 1024), f16 out
  gemm_f16_kernel<true>
      <<<dim3(NQKV / 128, MROWS / 64), 256, 0, stream>>>(
          x16, wqkv16, qkv, MROWS, NQKV, DMODEL);

  // 2) RoPE + repack
  rope_pack_kernel<<<(BATCH * NH * SEQ * 32) / 256, 256, 0, stream>>>(
      qkv, cosT, sinT, Qp, Ktw, Vp);

  // 3) flash attention
  attn_kernel<<<dim3(SEQ / 64, BATCH * NH), 128, 0, stream>>>(Qp, Ktw, Vp, Og);

  // 4) out = O @ w_out   (4096 x 1024 x 1024), fp32 out
  gemm_f16_kernel<false>
      <<<dim3(DMODEL / 128, MROWS / 64), 256, 0, stream>>>(
          Og, wout16, d_out, MROWS, DMODEL, DMODEL);
}
